// SparseNet_60559038874122
// MI455X (gfx1250) — matrix operand order repaired and checked by reading, not compiled, not run on hardware
//
#include <hip/hip_runtime.h>

// ---------------------------------------------------------------------------
// MI455X / gfx1250 implementation.
// Live math (reference's `v` is dead: out = softmax(q k^T * s) @ k):
//   1) qk = x @ w_qkv[0:512]^T + b -> RoPE -> scale folded into q (WMMA GEMM),
//      writing q, k (row-major) and kT (d-major) f16 workspaces.
//   2) per (b,head): flash attention; each 128-thread block shares one (b,head),
//      k/kT chunks are staged into LDS once per block via the Tensor Data Mover
//      (tensor_load_to_lds + s_wait_tensorcnt), double buffered, 4 WMMAs/chunk.
// Whole working set (~15MB) fits in L2 -> matrix-engine bound, not HBM bound.
// ---------------------------------------------------------------------------

typedef __attribute__((ext_vector_type(16))) _Float16     v16h;
typedef __attribute__((ext_vector_type(8)))  float        v8f;
typedef __attribute__((ext_vector_type(4)))  unsigned int u32x4;
typedef __attribute__((ext_vector_type(8)))  int          i32x8;
typedef __attribute__((ext_vector_type(4)))  int          i32x4;

#define BB      4
#define HH      48
#define WW      48
#define DIMC    256
#define NHEADS  8
#define DHD     32
#define SS      (HH * WW)          // 2304 tokens per image
#define NQK     512                // q + k output features (v skipped: dead)
#define NCHUNK  (SS / 32)          // 72 key chunks
#define QK_SCALE 0.17677669529663687f   // 32^-0.5

// A-fragment K index, contiguous k order used by this file (B loads match):
//   VGPR pair v, lane-half h: K = 16*h + 2*v
__device__ __forceinline__ int a_frag_k(int v, int half) {
    return half * 16 + v * 2;
}

// TDM 2-D tile load: D# per CDNA5 ISA §8.3/8.4 (group0: count=1, lds_addr,
// 57-bit global addr, type=2; group1: data_size=2B, dims/tile/stride).
// This toolchain exposes the 6-arg builtin: (g0, g1, g2, g3, g4, cpol);
// groups 2/3 (+spare group operand) are zero for 2-D tensors.
__device__ __forceinline__ void tdm_load_2d(unsigned lds_off, const void* gptr,
                                            unsigned td0, unsigned td1,
                                            unsigned tile0, unsigned tile1,
                                            unsigned stride0)
{
    unsigned long long ga = (unsigned long long)(uintptr_t)gptr;
    u32x4 g0;
    g0[0] = 1u;                                              // count=1, user D#
    g0[1] = lds_off;                                         // LDS byte offset
    g0[2] = (unsigned)ga;                                    // global_addr lo
    g0[3] = (unsigned)((ga >> 32) & 0x01FFFFFFu)             // global_addr hi
          | 0x80000000u;                                     // type=2 (image)
    i32x8 g1;
    g1[0] = (int)(1u << 16);                                 // data_size=1 (2B)
    g1[1] = (int)((td0 & 0xFFFFu) << 16);                    // tensor_dim0 lo
    g1[2] = (int)((td0 >> 16) | ((td1 & 0xFFFFu) << 16));    // td0 hi | td1 lo
    g1[3] = (int)((td1 >> 16) | (tile0 << 16));              // td1 hi | tile0
    g1[4] = (int)(tile1 & 0xFFFFu);                          // tile1, tile2=0
    g1[5] = (int)stride0;                                    // dim0 stride lo
    g1[6] = 0;                                               // stride hi, d1s lo
    g1[7] = 0;
    i32x4 z4 = {};
    i32x8 z8 = {};
    __builtin_amdgcn_tensor_load_to_lds(g0, g1, z4, z4, z8, 0);
}

// ============================================================================
// Kernel 1: fused QK projection GEMM + bias + RoPE (+ scale folded into q).
// One wave computes one 16x16 tile of the (9216 x 512) output.
// ============================================================================
__global__ __launch_bounds__(128) void qk_rope_gemm(
    const float* __restrict__ x,     // (B,H,W,DIMC) row-major
    const float* __restrict__ w,     // (768, 256); we use rows [0,512)
    const float* __restrict__ bias,  // (768)
    const float* __restrict__ sinb,  // (SS,DHD)
    const float* __restrict__ cosb,  // (SS,DHD)
    _Float16* __restrict__ qws,      // (B*NHEADS, SS, DHD) f16, scaled
    _Float16* __restrict__ kws,      // (B*NHEADS, SS, DHD) f16
    _Float16* __restrict__ ktws)     // (B*NHEADS, DHD, SS) f16 (transposed k)
{
    const int lane = threadIdx.x & 31;
    const int wave = threadIdx.x >> 5;
    const int tile = blockIdx.x * 4 + wave;        // 576*32 = 18432 tiles
    const int tm   = tile / (NQK / 16);            // token-tile
    const int tn   = tile % (NQK / 16);            // feature-tile

    const int mloc = lane & 15;
    const int half = lane >> 4;

    const int arow = tm * 16 + mloc;               // token row for A loads
    const int ocol = tn * 16 + mloc;               // feature col for B loads

    v8f acc = {};
    // K = 256 in 8 steps of 32 -> 8 back-to-back v_wmma_f32_16x16x32_f16
#pragma unroll
    for (int kb = 0; kb < DIMC; kb += 32) {
        v16h a, b;
        const float* xp = x + arow * DIMC + kb;
        const float* wp = w + ocol * DIMC + kb + half * 16;
#pragma unroll
        for (int v = 0; v < 8; ++v) {
            const int kk = a_frag_k(v, half);
            a[2 * v]     = (_Float16)xp[kk];
            a[2 * v + 1] = (_Float16)xp[kk + 1];
            b[2 * v]     = (_Float16)wp[2 * v];
            b[2 * v + 1] = (_Float16)wp[2 * v + 1];
        }
        acc = __builtin_amdgcn_wmma_f32_16x16x32_f16(false, a, false, b,
                                                     (short)0, acc, false, false);
    }

    // Epilogue: bias + RoPE. C-frag: row m = v + 8*half, col n = lane&15.
    // RoPE partner (feature o^1) sits in the adjacent lane -> shfl_xor(1).
#pragma unroll
    for (int v = 0; v < 8; ++v) {
        const int m   = v + 8 * half;
        const int row = tm * 16 + m;
        const int o   = tn * 16 + mloc;
        float val = acc[v] + bias[o];
        float pv  = __shfl_xor(val, 1, 32);   // partner feature o^1, same row

        const int d    = o & 31;
        const int head = (o >> 5) & 7;
        const int isK  = o >> 8;              // uniform per wave (tn-block)
        const int s    = row % SS;
        const int bi   = row / SS;
        const int bh   = bi * NHEADS + head;

        const float cs = cosb[s * DHD + d];
        const float sn = sinb[s * DHD + d];
        // y[2i] = x[2i]*cos - x[2i+1]*sin ; y[2i+1] = x[2i+1]*cos + x[2i]*sin
        const float y = (d & 1) ? (val * cs + pv * sn) : (val * cs - pv * sn);

        if (isK) {
            const _Float16 hy = (_Float16)y;
            kws [((size_t)bh * SS + s) * DHD + d] = hy;       // [s][d]
            ktws[((size_t)bh * DHD + d) * SS + s] = hy;       // [d][s]
        } else {
            qws [((size_t)bh * SS + s) * DHD + d] = (_Float16)(y * QK_SCALE);
        }
    }
}

// ============================================================================
// Kernel 2: flash attention, out = softmax(q k^T) @ k  (all-f16 WMMA, f32 acc)
// Block = 4 waves sharing one (b,head); each wave owns 16 query rows.
// K chunks (32 keys) staged via TDM into double-buffered LDS.
// ============================================================================
__global__ __launch_bounds__(128) void flash_attn(
    const _Float16* __restrict__ qws,
    const _Float16* __restrict__ kws,
    const _Float16* __restrict__ ktws,
    float* __restrict__ out)             // (B,H,W,DIMC) fp32
{
    __shared__ __align__(64) _Float16 kbuf [2][32 * DHD];  // [key][d] chunk
    __shared__ __align__(64) _Float16 ktbuf[2][DHD * 32];  // [d][key] chunk
    __shared__ __align__(64) _Float16 pl[4][16 * 32];      // per-wave P tile

    const int lane    = threadIdx.x & 31;
    const int wave    = threadIdx.x >> 5;
    const int tile    = blockIdx.x * 4 + wave;   // 32 * 144 = 4608 tiles
    const int bh      = tile / (SS / 16);        // same for all 4 waves (144%4==0)
    const int rowtile = tile % (SS / 16);
    const int bi      = bh / NHEADS;
    const int head    = bh % NHEADS;

    const int mloc = lane & 15;
    const int half = lane >> 4;

    // Q A-fragment (16x32), scale already folded in.
    v16h qa;
    {
        const _Float16* qp = qws + ((size_t)bh * SS + rowtile * 16 + mloc) * DHD;
#pragma unroll
        for (int v = 0; v < 8; ++v) {
            const int kk = a_frag_k(v, half);
            qa[2 * v]     = qp[kk];
            qa[2 * v + 1] = qp[kk + 1];
        }
    }

    v8f o0 = {}, o1 = {};                 // output acc, d=[0,16) and [16,32)
    float rmax[8], rsum[8];
#pragma unroll
    for (int v = 0; v < 8; ++v) { rmax[v] = -3.0e38f; rsum[v] = 0.0f; }

    const _Float16* kg  = kws  + (size_t)bh * SS * DHD;   // [s][d]
    const _Float16* ktg = ktws + (size_t)bh * DHD * SS;   // [d][s]
    _Float16* myp = pl[wave];

    // Prologue: wave0 issues TDM for chunk 0 into buffer 0.
    if (wave == 0) {
        tdm_load_2d((unsigned)(uintptr_t)&kbuf[0][0],  kg,  DHD, SS, 32, 32, DHD);
        tdm_load_2d((unsigned)(uintptr_t)&ktbuf[0][0], ktg, SS, DHD, 32, 32, SS);
    }

    for (int c = 0; c < NCHUNK; ++c) {
        const int cur = c & 1;
        if (wave == 0) __builtin_amdgcn_s_wait_tensorcnt(0);  // chunk c landed
        __syncthreads();   // data visible to all waves; prev compute done
        if (wave == 0 && c + 1 < NCHUNK) {
            const int jn = (c + 1) * 32;
            tdm_load_2d((unsigned)(uintptr_t)&kbuf[cur ^ 1][0],
                        kg + (size_t)jn * DHD, DHD, SS, 32, 32, DHD);
            tdm_load_2d((unsigned)(uintptr_t)&ktbuf[cur ^ 1][0],
                        ktg + jn, SS, DHD, 32, 32, SS);
        }

        // Score B-frags from LDS: lane = key column, lane-half = dh-half.
        const _Float16* kb = kbuf[cur];
        const v16h kb0 = *(const v16h*)(kb + mloc * DHD + half * 16);
        const v16h kb1 = *(const v16h*)(kb + (16 + mloc) * DHD + half * 16);
        const v8f zz = {};
        v8f s0 = __builtin_amdgcn_wmma_f32_16x16x32_f16(false, qa, false, kb0,
                                                        (short)0, zz, false, false);
        v8f s1 = __builtin_amdgcn_wmma_f32_16x16x32_f16(false, qa, false, kb1,
                                                        (short)0, zz, false, false);

        // Online softmax. Row m = v + 8*half lives in one 16-lane half:
        // xor-reduce over masks 1,2,4,8 stays inside the half.
#pragma unroll
        for (int v = 0; v < 8; ++v) {
            float tmv = fmaxf(s0[v], s1[v]);
            tmv = fmaxf(tmv, __shfl_xor(tmv, 1, 32));
            tmv = fmaxf(tmv, __shfl_xor(tmv, 2, 32));
            tmv = fmaxf(tmv, __shfl_xor(tmv, 4, 32));
            tmv = fmaxf(tmv, __shfl_xor(tmv, 8, 32));
            const float nm   = fmaxf(rmax[v], tmv);
            const float corr = __expf(rmax[v] - nm);
            const float p0   = __expf(s0[v] - nm);
            const float p1   = __expf(s1[v] - nm);
            float ps = p0 + p1;
            ps += __shfl_xor(ps, 1, 32);
            ps += __shfl_xor(ps, 2, 32);
            ps += __shfl_xor(ps, 4, 32);
            ps += __shfl_xor(ps, 8, 32);
            rsum[v] = rsum[v] * corr + ps;
            rmax[v] = nm;
            o0[v] *= corr;
            o1[v] *= corr;
            const int m = v + 8 * half;
            myp[m * 32 + mloc]      = (_Float16)p0;   // C-layout -> LDS
            myp[m * 32 + 16 + mloc] = (_Float16)p1;
        }

        // Re-read P in A-layout (same-wave LDS is in-order).
        v16h pa;
#pragma unroll
        for (int v = 0; v < 8; ++v) {
            const int kk = a_frag_k(v, half);
            pa[2 * v]     = myp[mloc * 32 + kk];
            pa[2 * v + 1] = myp[mloc * 32 + kk + 1];
        }

        // P@K B-frags from transposed LDS chunk: lane = d column, half = j-half;
        // each lane reads 16 contiguous halfs (32B) -> ds_load_b128 pairs.
        const _Float16* kt = ktbuf[cur];
        const v16h b0 = *(const v16h*)(kt + mloc * 32 + half * 16);
        const v16h b1 = *(const v16h*)(kt + (mloc + 16) * 32 + half * 16);
        o0 = __builtin_amdgcn_wmma_f32_16x16x32_f16(false, pa, false, b0,
                                                    (short)0, o0, false, false);
        o1 = __builtin_amdgcn_wmma_f32_16x16x32_f16(false, pa, false, b1,
                                                    (short)0, o1, false, false);
    }

    // Normalize + store fp32 output, channel = head*32 + d.
#pragma unroll
    for (int v = 0; v < 8; ++v) {
        const int m = v + 8 * half;
        const int s = rowtile * 16 + m;
        const float inv = 1.0f / rsum[v];
        float* op = out + ((size_t)bi * SS + s) * DIMC + head * DHD;
        op[mloc]      = o0[v] * inv;
        op[16 + mloc] = o1[v] * inv;
    }
}

// ============================================================================
extern "C" void kernel_launch(void* const* d_in, const int* in_sizes, int n_in,
                              void* d_out, int out_size, void* d_ws, size_t ws_size,
                              hipStream_t stream)
{
    (void)in_sizes; (void)n_in; (void)out_size; (void)ws_size;
    const float* x    = (const float*)d_in[0];
    const float* wq   = (const float*)d_in[1];
    const float* bq   = (const float*)d_in[2];
    const float* sinb = (const float*)d_in[3];
    const float* cosb = (const float*)d_in[4];
    float* out        = (float*)d_out;

    const size_t nkv = (size_t)BB * NHEADS * SS * DHD;   // 2,359,296 halfs
    _Float16* qws  = (_Float16*)d_ws;        // 4.72 MB
    _Float16* kws  = qws + nkv;              // 4.72 MB
    _Float16* ktws = kws + nkv;              // 4.72 MB (transposed k)

    // Kernel 1: (9216/16)*(512/16) = 18432 wave-tiles, 4 waves/block.
    qk_rope_gemm<<<18432 / 4, 128, 0, stream>>>(x, wq, bq, sinb, cosb,
                                                qws, kws, ktws);

    // Kernel 2: 32 (b,h) * 144 row-tiles = 4608 wave-tiles, 4 waves/block.
    flash_attn<<<4608 / 4, 128, 0, stream>>>(qws, kws, ktws, out);
}
